// net_crossInteraction_66271345377782
// MI455X (gfx1250) — matrix operand order repaired and checked by reading, not compiled, not run on hardware
//
#include <hip/hip_runtime.h>
#include <hip/hip_bf16.h>
#include <math.h>

typedef __attribute__((ext_vector_type(16))) _Float16 v16h;
typedef __attribute__((ext_vector_type(8)))  _Float16 v8h;
typedef __attribute__((ext_vector_type(8)))  float    v8f;

#define BATCH 4
#define SEQ   256   // L (acts as conv channel axis)
#define DIM   256   // D (acts as conv length axis)
#define CH    256
#define LPAD  8
#define LROWS 272   // 256 + 2*LPAD guard rows (max conv reach is -5..+6)

#define WMMA_F16(a, b, c) \
  __builtin_amdgcn_wmma_f32_16x16x32_f16(false, (a), false, (b), (short)0, (c), false, false)

// ---------------- WMMA fragment loaders (wave32, 16x16x32 f16) ----------------
// A 16x32 f16: lane<16 holds row m=lane, K[kb..kb+15];
// lane>=16 holds row m=lane-16, K[kb+16..kb+31].
__device__ __forceinline__ v16h load_a16(const _Float16* __restrict__ A, int lda,
                                         int m0, int kb, int lane) {
  int row = m0 + (lane & 15);
  int k0  = kb + ((lane >> 4) << 4);
  const _Float16* p = A + (size_t)row * lda + k0;
  v8h lo = *(const v8h*)p;
  v8h hi = *(const v8h*)(p + 8);
  v16h r;
#pragma unroll
  for (int j = 0; j < 8; ++j) { r[j] = lo[j]; r[j + 8] = hi[j]; }
  return r;
}

// B 32x16 f16 where B[K][n] = R[n][K] with R row-major (ldr): lane = column n,
// 16 contiguous halves over K; lanes>=16 take K+16.
__device__ __forceinline__ v16h load_b16(const _Float16* __restrict__ R, int ldr,
                                         int n0, int kb, int lane) {
  int col = n0 + (lane & 15);
  int k0  = kb + ((lane >> 4) << 4);
  return *(const v16h*)(R + (size_t)col * ldr + k0);
}

// ---------------- init / prep kernels ----------------
__global__ void zero_half_kernel(_Float16* p, size_t n) {
  size_t i = (size_t)blockIdx.x * blockDim.x + threadIdx.x;
  size_t stride = (size_t)gridDim.x * blockDim.x;
  for (; i < n; i += stride) p[i] = (_Float16)0.f;
}

__global__ void embed_kernel(const int* __restrict__ tok, const float* __restrict__ emb,
                             _Float16* __restrict__ Xa) {
  int idx = blockIdx.x * blockDim.x + threadIdx.x;  // B*L*D = 262144
  int i = idx & 255;            // token position = conv channel
  int d = (idx >> 8) & 255;     // embedding dim = conv time
  int b = idx >> 16;
  int t = tok[b * SEQ + i];
  Xa[(size_t)b * LROWS * CH + (size_t)(LPAD + d) * CH + i] = (_Float16)emb[t * DIM + d];
}

// w (O=256, I=256, taps) f32 -> Wt (tap, o, i) f16
__global__ void prep_convw_kernel(const float* __restrict__ w, _Float16* __restrict__ Wt,
                                  int taps) {
  int idx = blockIdx.x * blockDim.x + threadIdx.x;  // 65536*taps
  int i = idx & 255;
  int o = (idx >> 8) & 255;
  int tap = idx >> 16;
  Wt[(size_t)tap * CH * CH + (size_t)o * CH + i] =
      (_Float16)w[(size_t)o * CH * taps + (size_t)i * taps + tap];
}

// ja (j, d) f32 -> jaT (d, j) f16
__global__ void prep_jaw_kernel(const float* __restrict__ w, _Float16* __restrict__ wt) {
  int idx = blockIdx.x * blockDim.x + threadIdx.x;  // 65536
  int j = idx & 255;
  int d = idx >> 8;
  wt[(size_t)d * 256 + j] = (_Float16)w[(size_t)j * 256 + d];
}

// ------ conv as tap-summed GEMM, 2x2 register-blocked (32x32 C per wave) ------
// Y[o,t] = relu(b[o] + sum_k W_k @ X[.,t+k-padL])
__global__ void conv_gemm_kernel(const _Float16* __restrict__ Xin,  // (B,272,256) time-major f16
                                 const _Float16* __restrict__ Wt,   // (taps,256,256)
                                 const float* __restrict__ bias,
                                 int taps, int padL,
                                 _Float16* __restrict__ Xout,       // (B,272,256) time-major f16
                                 _Float16* __restrict__ Ec) {       // optional (B,256,256) ch-major
  int lane = threadIdx.x & 31;
  int tile = blockIdx.x;                 // B*8*8 = 256
  int b  = tile >> 6;
  int m0 = ((tile >> 3) & 7) << 5;       // out-channel tile (32 rows)
  int t0 = (tile & 7) << 5;              // time tile (32 cols)
  const _Float16* Xb = Xin + (size_t)b * LROWS * CH;
  v8f acc00 = {}, acc01 = {}, acc10 = {}, acc11 = {};
  for (int tap = 0; tap < taps; ++tap) {
    const _Float16* Wtap = Wt + (size_t)tap * CH * CH;
    int shift = tap - padL;
    const _Float16* Brow0 =
        Xb + (size_t)(LPAD + t0 + (lane & 15) + shift) * CH + ((lane >> 4) << 4);
    const _Float16* Brow1 = Brow0 + (size_t)16 * CH;
#pragma unroll
    for (int kc = 0; kc < CH; kc += 32) {
      v16h a0 = load_a16(Wtap, CH, m0, kc, lane);
      v16h a1 = load_a16(Wtap, CH, m0 + 16, kc, lane);
      v16h b0 = *(const v16h*)(Brow0 + kc);
      v16h b1 = *(const v16h*)(Brow1 + kc);
      acc00 = WMMA_F16(a0, b0, acc00);
      acc01 = WMMA_F16(a0, b1, acc01);
      acc10 = WMMA_F16(a1, b0, acc10);
      acc11 = WMMA_F16(a1, b1, acc11);
    }
  }
  v8f* accs[2][2] = {{&acc00, &acc01}, {&acc10, &acc11}};
#pragma unroll
  for (int mi = 0; mi < 2; ++mi) {
    int rb = m0 + mi * 16 + ((lane >> 4) << 3);
#pragma unroll
    for (int ni = 0; ni < 2; ++ni) {
      int t = t0 + ni * 16 + (lane & 15);
      v8f c = *accs[mi][ni];
      v8h outv;
#pragma unroll
      for (int j = 0; j < 8; ++j) {
        float v = c[j] + bias[rb + j];
        v = v > 0.f ? v : 0.f;           // relu
        outv[j] = (_Float16)v;
      }
      *(v8h*)(Xout + (size_t)b * LROWS * CH + (size_t)(LPAD + t) * CH + rb) = outv;
      if (Ec) {
#pragma unroll
        for (int j = 0; j < 8; ++j)
          Ec[(size_t)b * CH * CH + (size_t)(rb + j) * CH + t] = outv[j];
      }
    }
  }
}

// ------ q/k GEMM, 2x2 blocked: Q[i,d] = Ec[i,:] @ ja_w[:,d] + b[d] ------
__global__ void qk_gemm_kernel(const _Float16* __restrict__ Ec,   // (B,256i,256j)
                               const _Float16* __restrict__ WT,   // (256d,256j) = ja_w^T
                               const float* __restrict__ bias,
                               _Float16* __restrict__ Q) {        // (B,256i,256d)
  int lane = threadIdx.x & 31;
  int tile = blockIdx.x;                 // 256
  int b  = tile >> 6;
  int m0 = ((tile >> 3) & 7) << 5;       // i tile
  int n0 = (tile & 7) << 5;              // d tile
  const _Float16* A = Ec + (size_t)b * CH * CH;
  v8f acc00 = {}, acc01 = {}, acc10 = {}, acc11 = {};
#pragma unroll
  for (int kc = 0; kc < CH; kc += 32) {
    v16h a0 = load_a16(A, CH, m0, kc, lane);
    v16h a1 = load_a16(A, CH, m0 + 16, kc, lane);
    v16h b0 = load_b16(WT, CH, n0, kc, lane);
    v16h b1 = load_b16(WT, CH, n0 + 16, kc, lane);
    acc00 = WMMA_F16(a0, b0, acc00);
    acc01 = WMMA_F16(a0, b1, acc01);
    acc10 = WMMA_F16(a1, b0, acc10);
    acc11 = WMMA_F16(a1, b1, acc11);
  }
  v8f* accs[2][2] = {{&acc00, &acc01}, {&acc10, &acc11}};
#pragma unroll
  for (int mi = 0; mi < 2; ++mi) {
    int rb = m0 + mi * 16 + ((lane >> 4) << 3);
#pragma unroll
    for (int ni = 0; ni < 2; ++ni) {
      int d = n0 + ni * 16 + (lane & 15);
      float bv = bias[d];
      v8f c = *accs[mi][ni];
#pragma unroll
      for (int j = 0; j < 8; ++j)
        Q[(size_t)b * CH * CH + (size_t)(rb + j) * CH + d] = (_Float16)(c[j] + bv);
    }
  }
}

// ------ inter GEMM, 2x2 blocked: sig[i,k] = sigmoid(q[i,:]·k[k,:]); per-tile sums ------
__global__ void inter_gemm_kernel(const _Float16* __restrict__ Q,
                                  const _Float16* __restrict__ K,
                                  float* __restrict__ sig, float* __restrict__ tilesum) {
  int lane = threadIdx.x & 31;
  int tile = blockIdx.x;                 // 256 (64 per batch)
  int b  = tile >> 6;
  int m0 = ((tile >> 3) & 7) << 5;       // i tile
  int n0 = (tile & 7) << 5;              // k tile
  const _Float16* Qa = Q + (size_t)b * CH * CH;
  const _Float16* Kb = K + (size_t)b * CH * CH;
  v8f acc00 = {}, acc01 = {}, acc10 = {}, acc11 = {};
#pragma unroll
  for (int kc = 0; kc < CH; kc += 32) {
    v16h a0 = load_a16(Qa, CH, m0, kc, lane);
    v16h a1 = load_a16(Qa, CH, m0 + 16, kc, lane);
    v16h b0 = load_b16(Kb, CH, n0, kc, lane);       // B[K=d][n]=K[n][d]: no transpose
    v16h b1 = load_b16(Kb, CH, n0 + 16, kc, lane);
    acc00 = WMMA_F16(a0, b0, acc00);
    acc01 = WMMA_F16(a0, b1, acc01);
    acc10 = WMMA_F16(a1, b0, acc10);
    acc11 = WMMA_F16(a1, b1, acc11);
  }
  v8f* accs[2][2] = {{&acc00, &acc01}, {&acc10, &acc11}};
  float s8 = 0.f;
#pragma unroll
  for (int mi = 0; mi < 2; ++mi) {
    int rb = m0 + mi * 16 + ((lane >> 4) << 3);
#pragma unroll
    for (int ni = 0; ni < 2; ++ni) {
      int kcol = n0 + ni * 16 + (lane & 15);
      v8f c = *accs[mi][ni];
#pragma unroll
      for (int j = 0; j < 8; ++j) {
        float s = 1.f / (1.f + __expf(-c[j]));
        sig[(size_t)b * 65536 + (size_t)(rb + j) * 256 + kcol] = s;
        s8 += s;
      }
    }
  }
#pragma unroll
  for (int d = 16; d > 0; d >>= 1) s8 += __shfl_down(s8, d, 32);
  if (lane == 0) tilesum[tile] = s8;     // deterministic (no atomics)
}

__global__ void reduce_sum_kernel(const float* __restrict__ tilesum, float* __restrict__ bsum) {
  __shared__ float red[64];
  int tid = threadIdx.x;                 // 64 tiles per batch
  red[tid] = tilesum[blockIdx.x * 64 + tid];
  __syncthreads();
  for (int s = 32; s > 0; s >>= 1) {
    if (tid < s) red[tid] += red[tid + s];
    __syncthreads();
  }
  if (tid == 0) bsum[blockIdx.x] = red[0];
}

__global__ void normalize_kernel(const float* __restrict__ sig, const float* __restrict__ bsum,
                                 float* __restrict__ out) {
  int idx = blockIdx.x * blockDim.x + threadIdx.x;  // B*L*L
  out[idx] = sig[idx] / bsum[idx >> 16];
}

// ---------------- pooled[b,j] = sum_{i,k} tanh(e1[i,j]*e2[k,j]) * inter[i,k] ----------
__global__ void pool_kernel(const _Float16* __restrict__ E1, const _Float16* __restrict__ E2,
                            const float* __restrict__ inter, float* __restrict__ pooled) {
  __shared__ float e2row[256];
  __shared__ float red[256];
  int b = blockIdx.x >> 8;
  int j = blockIdx.x & 255;
  int i = threadIdx.x;
  const _Float16* r1 = E1 + (size_t)b * LROWS * CH + (size_t)(LPAD + j) * CH;
  const _Float16* r2 = E2 + (size_t)b * LROWS * CH + (size_t)(LPAD + j) * CH;
  e2row[i] = (float)r2[i];
  __syncthreads();
  float a1 = (float)r1[i];
  const float* wrow = inter + (size_t)b * 65536 + (size_t)i * 256;
  float acc = 0.f;
  for (int k = 0; k < 256; ++k) acc += tanhf(a1 * e2row[k]) * wrow[k];
  red[i] = acc;
  __syncthreads();
  for (int s = 128; s > 0; s >>= 1) {
    if (i < s) red[i] += red[i + s];
    __syncthreads();
  }
  if (i == 0) pooled[b * DIM + j] = red[0];
}

// ---------------- tiny head: conv(64,1,4,s2,p1)+lrelu -> maxpool4 -> fc 2048/600/300/1 --
__global__ void head_kernel(const float* __restrict__ pooled,
                            const float* __restrict__ rcw, const float* __restrict__ rcb,
                            const float* __restrict__ f1w, const float* __restrict__ f1b,
                            const float* __restrict__ f2w, const float* __restrict__ f2b,
                            const float* __restrict__ f3w, const float* __restrict__ f3b,
                            float* __restrict__ affn) {
  __shared__ float ph[256];
  __shared__ float hh[2048];
  __shared__ float h1[600];
  __shared__ float h2[300];
  __shared__ float red[256];
  int b = blockIdx.x, tid = threadIdx.x;
  ph[tid] = pooled[b * 256 + tid];
  __syncthreads();
  for (int idx = tid; idx < 2048; idx += 256) {
    int ch = idx >> 5, m = idx & 31;
    float mx = -1e30f;
    for (int p = 0; p < 4; ++p) {
      int t = m * 4 + p;                      // 0..127
      float s = rcb[ch];
      for (int kk = 0; kk < 4; ++kk) {
        int pos = t * 2 + kk - 1;             // stride 2, pad (1,1)
        if (pos >= 0 && pos < 256) s += ph[pos] * rcw[ch * 4 + kk];
      }
      s = s >= 0.f ? s : 0.1f * s;
      mx = fmaxf(mx, s);
    }
    hh[idx] = mx;
  }
  __syncthreads();
  for (int o = tid; o < 600; o += 256) {
    float s = f1b[o];
    for (int x = 0; x < 2048; ++x) s += hh[x] * f1w[(size_t)x * 600 + o];
    h1[o] = s >= 0.f ? s : 0.1f * s;
  }
  __syncthreads();
  for (int o = tid; o < 300; o += 256) {
    float s = f2b[o];
    for (int x = 0; x < 600; ++x) s += h1[x] * f2w[x * 300 + o];
    h2[o] = s >= 0.f ? s : 0.1f * s;
  }
  __syncthreads();
  float acc = 0.f;
  for (int x = tid; x < 300; x += 256) acc += h2[x] * f3w[x];
  red[tid] = acc;
  __syncthreads();
  for (int s = 128; s > 0; s >>= 1) {
    if (tid < s) red[tid] += red[tid + s];
    __syncthreads();
  }
  if (tid == 0) affn[b] = red[0] + f3b[0];
}

// ---------------- host ----------------
extern "C" void kernel_launch(void* const* d_in, const int* in_sizes, int n_in,
                              void* d_out, int out_size, void* d_ws, size_t ws_size,
                              hipStream_t stream) {
  const int*   tok1 = (const int*)d_in[0];
  const int*   tok2 = (const int*)d_in[1];
  const float* emb  = (const float*)d_in[2];
  const float* w0 = (const float*)d_in[3];   const float* b0 = (const float*)d_in[4];
  const float* w1 = (const float*)d_in[5];   const float* b1 = (const float*)d_in[6];
  const float* w2 = (const float*)d_in[7];   const float* b2 = (const float*)d_in[8];
  const float* ja1w = (const float*)d_in[9];  const float* ja1b = (const float*)d_in[10];
  const float* ja2w = (const float*)d_in[11]; const float* ja2b = (const float*)d_in[12];
  const float* rcw = (const float*)d_in[13]; const float* rcb = (const float*)d_in[14];
  const float* f1w = (const float*)d_in[15]; const float* f1b = (const float*)d_in[16];
  const float* f2w = (const float*)d_in[17]; const float* f2b = (const float*)d_in[18];
  const float* f3w = (const float*)d_in[19]; const float* f3b = (const float*)d_in[20];
  float* out = (float*)d_out;

  char* ws = (char*)d_ws;
  size_t off = 0;
  auto alloc = [&](size_t bytes) -> char* {
    char* p = ws + off;
    off += (bytes + 255) & ~(size_t)255;
    return p;
  };

  const size_t PADHALF = (size_t)BATCH * LROWS * CH;   // halves per padded act buffer
  // 8 contiguous padded activation buffers (each a multiple of 256 B): one zero pass.
  _Float16* Xa1 = (_Float16*)alloc(PADHALF * 2);
  _Float16* Xb1 = (_Float16*)alloc(PADHALF * 2);
  _Float16* Xc1 = (_Float16*)alloc(PADHALF * 2);
  _Float16* Ep1 = (_Float16*)alloc(PADHALF * 2);
  _Float16* Xa2 = (_Float16*)alloc(PADHALF * 2);
  _Float16* Xb2 = (_Float16*)alloc(PADHALF * 2);
  _Float16* Xc2 = (_Float16*)alloc(PADHALF * 2);
  _Float16* Ep2 = (_Float16*)alloc(PADHALF * 2);
  _Float16* Ec1 = (_Float16*)alloc((size_t)BATCH * CH * CH * 2);
  _Float16* Ec2 = (_Float16*)alloc((size_t)BATCH * CH * CH * 2);
  _Float16* Wt0 = (_Float16*)alloc((size_t)4  * CH * CH * 2);
  _Float16* Wt1 = (_Float16*)alloc((size_t)8  * CH * CH * 2);
  _Float16* Wt2 = (_Float16*)alloc((size_t)12 * CH * CH * 2);
  _Float16* jaT1 = (_Float16*)alloc((size_t)CH * CH * 2);
  _Float16* jaT2 = (_Float16*)alloc((size_t)CH * CH * 2);
  _Float16* Qb = (_Float16*)alloc((size_t)BATCH * CH * CH * 2);
  _Float16* Kb = (_Float16*)alloc((size_t)BATCH * CH * CH * 2);
  float* sig     = (float*)alloc((size_t)BATCH * SEQ * SEQ * 4);
  float* tilesum = (float*)alloc(256 * 4);
  float* bsum    = (float*)alloc(16);
  float* pooled  = (float*)alloc((size_t)BATCH * DIM * 4);

  // 1) zero all padded activation buffers (guard rows must be 0 for conv boundary taps)
  zero_half_kernel<<<1024, 256, 0, stream>>>(Xa1, PADHALF * 8);
  // 2) embed both sequences into time-major f16
  embed_kernel<<<1024, 256, 0, stream>>>(tok1, emb, Xa1);
  embed_kernel<<<1024, 256, 0, stream>>>(tok2, emb, Xa2);
  // 3) weight prep (f32 -> f16, WMMA-friendly layouts)
  prep_convw_kernel<<<256 * 4,  256, 0, stream>>>(w0, Wt0, 4);
  prep_convw_kernel<<<256 * 8,  256, 0, stream>>>(w1, Wt1, 8);
  prep_convw_kernel<<<256 * 12, 256, 0, stream>>>(w2, Wt2, 12);
  prep_jaw_kernel<<<256, 256, 0, stream>>>(ja1w, jaT1);
  prep_jaw_kernel<<<256, 256, 0, stream>>>(ja2w, jaT2);
  // 4) encoder conv stacks (WMMA, 2x2 register-blocked), pads: (1,2) (3,4) (5,6)
  conv_gemm_kernel<<<256, 32, 0, stream>>>(Xa1, Wt0, b0, 4, 1, Xb1, nullptr);
  conv_gemm_kernel<<<256, 32, 0, stream>>>(Xb1, Wt1, b1, 8, 3, Xc1, nullptr);
  conv_gemm_kernel<<<256, 32, 0, stream>>>(Xc1, Wt2, b2, 12, 5, Ep1, Ec1);
  conv_gemm_kernel<<<256, 32, 0, stream>>>(Xa2, Wt0, b0, 4, 1, Xb2, nullptr);
  conv_gemm_kernel<<<256, 32, 0, stream>>>(Xb2, Wt1, b1, 8, 3, Xc2, nullptr);
  conv_gemm_kernel<<<256, 32, 0, stream>>>(Xc2, Wt2, b2, 12, 5, Ep2, Ec2);
  // 5) joint-attention projections (WMMA, 2x2 blocked)
  qk_gemm_kernel<<<256, 32, 0, stream>>>(Ec1, jaT1, ja1b, Qb);
  qk_gemm_kernel<<<256, 32, 0, stream>>>(Ec2, jaT2, ja2b, Kb);
  // 6) inter = sigmoid(q @ k^T) (WMMA) + deterministic batch sums + normalize to d_out
  inter_gemm_kernel<<<256, 32, 0, stream>>>(Qb, Kb, sig, tilesum);
  reduce_sum_kernel<<<BATCH, 64, 0, stream>>>(tilesum, bsum);
  normalize_kernel<<<1024, 256, 0, stream>>>(sig, bsum, out);
  // 7) bilinear tanh pooling weighted by normalized inter
  pool_kernel<<<1024, 256, 0, stream>>>(Ep1, Ep2, out, pooled);
  // 8) tiny conv/maxpool/MLP head -> affn at out[B*L*L .. +B)
  head_kernel<<<BATCH, 256, 0, stream>>>(pooled, rcw, rcb, f1w, f1b, f2w, f2b, f3w, f3b,
                                         out + (size_t)BATCH * SEQ * SEQ);
}
